// Scale_Attention_74947179315590
// MI455X (gfx1250) — matrix operand order repaired and checked by reading, not compiled, not run on hardware
//
#include <hip/hip_runtime.h>

typedef __bf16 bf16_t;
typedef bf16_t v16bf __attribute__((ext_vector_type(16)));
typedef bf16_t v8bf  __attribute__((ext_vector_type(8)));
typedef float  v8f   __attribute__((ext_vector_type(8)));

#define B_   2
#define CK   256
#define N_   512     // tokens per block (= k sequence length)
#define C_   32
#define NQ   32768
#define G_   64

// CDNA5 async global->LDS copy (ASYNCcnt-tracked). lds_off is the byte offset
// into the wave's LDS allocation (== dynamic-smem offset: no static LDS here).
__device__ __forceinline__ void async_ld_b128(unsigned lds_off, const void* gptr) {
  asm volatile("global_load_async_to_lds_b128 %0, %1, off"
               :: "v"(lds_off), "v"(gptr) : "memory");
}
__device__ __forceinline__ void wait_async0() {
  asm volatile("s_wait_asynccnt 0x0" ::: "memory");
}

// ---------------- kernel 1: kf[b,i,c] = sum_cc k[b,cc,i]*Wk[cc,c] + bk[c] (bf16 out) ----
__global__ void __launch_bounds__(512)
kf_proj_kernel(const float* __restrict__ k, const float* __restrict__ Wk,
               const float* __restrict__ bk, bf16_t* __restrict__ kf_ws) {
  __shared__ float wk_s[CK * C_];
  __shared__ float bk_s[C_];
  int tid = threadIdx.x;
  for (int t = tid; t < CK * C_; t += 512) wk_s[t] = Wk[t];
  if (tid < C_) bk_s[tid] = bk[tid];
  __syncthreads();
  int b = blockIdx.x;
  int i = tid;                         // 512 threads = 512 rows
  float acc[C_];
  #pragma unroll
  for (int c = 0; c < C_; ++c) acc[c] = bk_s[c];
  const float* kb = k + (size_t)b * CK * N_;
  for (int cc = 0; cc < CK; ++cc) {
    float kv = kb[cc * N_ + i];
    #pragma unroll
    for (int c = 0; c < C_; ++c) acc[c] += kv * wk_s[cc * C_ + c];
  }
  bf16_t* dst = kf_ws + ((size_t)b * N_ + i) * C_;
  #pragma unroll
  for (int c = 0; c < C_; ++c) dst[c] = (bf16_t)acc[c];
}

// ---------------- kernel 2: qb[b,g,j,c'] = sum_c q[b,c,pos]*Wq[c,c'] + bq (bf16, block layout) ----
__global__ void __launch_bounds__(256)
qb_proj_kernel(const float* __restrict__ q, const float* __restrict__ Wq,
               const float* __restrict__ bq, bf16_t* __restrict__ qb_ws) {
  __shared__ float wq_s[C_ * C_];
  __shared__ float bq_s[C_];
  int tid = threadIdx.x;
  for (int t = tid; t < C_ * C_; t += 256) wq_s[t] = Wq[t];
  if (tid < C_) bq_s[tid] = bq[tid];
  __syncthreads();
  int gidx = blockIdx.x * 256 + tid;   // 0..65535
  int b = gidx >> 15;
  int pos = gidx & (NQ - 1);
  float acc[C_];
  #pragma unroll
  for (int c = 0; c < C_; ++c) acc[c] = bq_s[c];
  const float* qp = q + (size_t)b * C_ * NQ + pos;
  for (int c = 0; c < C_; ++c) {
    float qv = qp[(size_t)c * NQ];
    #pragma unroll
    for (int c2 = 0; c2 < C_; ++c2) acc[c2] += qv * wq_s[c * C_ + c2];
  }
  int h = pos >> 10, w = (pos >> 5) & 31, d = pos & 31;
  int g = ((h >> 3) * 4 + (w >> 3)) * 4 + (d >> 3);
  int j = ((h & 7) * 8 + (w & 7)) * 8 + (d & 7);
  bf16_t* dst = qb_ws + (((size_t)b * G_ + g) * N_ + j) * C_;
  #pragma unroll
  for (int c = 0; c < C_; ++c) dst[c] = (bf16_t)acc[c];
}

// ---------------- kernel 3: per-(b,g) WMMA attention block ----------------
// LDS layout (bytes)
#define O_KF   0          // 512*32 bf16 = 32768
#define O_QB   32768      // 512*32 bf16 = 32768
#define O_S    65536      // 64*512 f32  = 131072 (scores; bf16 att overlaid in-place)
#define O_OUT  196608     // 512*32 f32  = 65536
#define O_WP   262144     // 32*32 f32   = 4096
#define O_BP   266240     // 128
#define O_RED  266368     // 1024
#define O_RED2 267392     // 1024
#define O_KN   268416     // 128
#define O_QN   268544     // 128
#define O_MN   268672     // 128
#define O_SC   268800     // 128
#define O_MISC 268928     // 64
#define SMEM_BYTES 268992

__global__ void __launch_bounds__(256)
attn_kernel(const bf16_t* __restrict__ kf_ws, const bf16_t* __restrict__ qb_ws,
            const float* __restrict__ Wp, const float* __restrict__ bp,
            float* __restrict__ out) {
  extern __shared__ char smem[];
  bf16_t* kf  = (bf16_t*)(smem + O_KF);
  bf16_t* qb  = (bf16_t*)(smem + O_QB);
  float*  S   = (float*)(smem + O_S);
  float*  ol  = (float*)(smem + O_OUT);
  float*  wp  = (float*)(smem + O_WP);
  float*  bps = (float*)(smem + O_BP);
  float*  red = (float*)(smem + O_RED);
  float*  red2= (float*)(smem + O_RED2);
  float*  kn  = (float*)(smem + O_KN);
  float*  qn  = (float*)(smem + O_QN);
  float*  mnv = (float*)(smem + O_MN);
  float*  scv = (float*)(smem + O_SC);
  float*  misc= (float*)(smem + O_MISC);

  int tid = threadIdx.x;
  int lane = tid & 31, w = tid >> 5;         // 8 waves
  int bb = blockIdx.x >> 6, g = blockIdx.x & 63;
  int nn = lane & 15, h = lane >> 4;         // WMMA lane decomposition

  // ---- async global->LDS tile loads (kf tile, qb tile), plus Wp/bp ----
  {
    const char* gk = (const char*)(kf_ws + (size_t)bb * N_ * C_);
    const char* gq = (const char*)(qb_ws + (((size_t)bb * G_ + g) * N_) * C_);
    #pragma unroll
    for (int t = 0; t < 8; ++t) {
      unsigned off = (unsigned)(tid + t * 256) * 16u;   // 2048 x 16B chunks each
      async_ld_b128(O_KF + off, gk + off);
      async_ld_b128(O_QB + off, gq + off);
    }
    for (int t = tid; t < 1024; t += 256) wp[t] = Wp[t];
    if (tid < 32) bps[tid] = bp[tid];
    wait_async0();
  }
  __syncthreads();

  // ---- per-channel norms over positions -> 1/denom scalar ----
  {
    float sk = 0.f, sq = 0.f;
    for (int i = w * 64; i < w * 64 + 64; ++i) {
      float a = (float)kf[i * C_ + lane]; sk += a * a;
      float b2 = (float)qb[i * C_ + lane]; sq += b2 * b2;
    }
    red[w * 32 + lane] = sk;
    red2[w * 32 + lane] = sq;
  }
  __syncthreads();
  if (tid < 32) {
    float sk = 0.f, sq = 0.f;
    for (int s2 = 0; s2 < 8; ++s2) { sk += red[s2 * 32 + tid]; sq += red2[s2 * 32 + tid]; }
    kn[tid] = sqrtf(sk); qn[tid] = sqrtf(sq);
  }
  __syncthreads();
  if (tid == 0) {
    float d = 0.f;
    for (int c = 0; c < 32; ++c) d += kn[c] * qn[c];
    misc[0] = 1.f / (d + 1e-4f);
  }
  __syncthreads();
  float inv_den = misc[0];

  // ---- hoist GEMM2 B fragments: B[K=i, N=c] = qb[i, ct*16+N]; strip-invariant ----
  int ct = w & 1;                      // c-tile owned by this wave in GEMM2
  v16bf bfrag2[16];
  #pragma unroll
  for (int s2 = 0; s2 < 16; ++s2) {
    v16bf bv;
    #pragma unroll
    for (int e = 0; e < 16; ++e)
      bv[e] = qb[(s2 * 32 + (e < 8 ? h * 8 + e : 16 + h * 8 + (e - 8))) * C_ + ct * 16 + nn];
    bfrag2[s2] = bv;
  }

  // ---- 8 strips of 64 j-rows each ----
  for (int strip = 0; strip < 8; ++strip) {
    int jbase = strip * 64;

    // GEMM1: S[j,i] = (qb[j,:] . kf[i,:]) * inv_den   (one 16x16x32 WMMA per tile)
    {
      int jt = (jbase >> 4) + (w >> 1);              // global j-tile of this wave
      const bf16_t* ap = qb + (jt * 16 + nn) * C_ + h * 8;
      v8bf alo = *(const v8bf*)ap;
      v8bf ahi = *(const v8bf*)(ap + 16);
      v16bf afrag = __builtin_shufflevector(alo, ahi,
          0,1,2,3,4,5,6,7,8,9,10,11,12,13,14,15);
      int itbase = (w & 1) * 16;
      for (int it = 0; it < 16; ++it) {
        int itg = itbase + it;
        const bf16_t* bpp = kf + (itg * 16 + nn) * C_ + h * 8;  // B[K,N] = kf[i,c]^T
        v8bf blo = *(const v8bf*)bpp;
        v8bf bhi = *(const v8bf*)(bpp + 16);
        v16bf bfrag = __builtin_shufflevector(blo, bhi,
            0,1,2,3,4,5,6,7,8,9,10,11,12,13,14,15);
        v8f acc = {};
        acc = __builtin_amdgcn_wmma_f32_16x16x32_bf16(
            false, afrag, false, bfrag, (short)0, acc, false, false);
        int srow = (w >> 1) * 16 + h * 8;            // strip-local
        #pragma unroll
        for (int r = 0; r < 8; ++r)
          S[(srow + r) * N_ + itg * 16 + nn] = acc[r] * inv_den;
      }
    }
    __syncthreads();

    // softmax over i (512) for 64 strip rows; wave w handles rows w*8..w*8+7.
    // Normalized result written back IN PLACE as bf16 into the low half of the row.
    for (int rr = 0; rr < 8; ++rr) {
      float* srow = S + (w * 8 + rr) * N_;
      bf16_t* arow = (bf16_t*)srow;                  // bf16 overlay (first 1KB of 2KB row)
      float vals[16];
      float m = -3.4e38f;
      #pragma unroll
      for (int t = 0; t < 16; ++t) { vals[t] = srow[lane + 32 * t]; m = fmaxf(m, vals[t]); }
      #pragma unroll
      for (int off = 16; off > 0; off >>= 1) m = fmaxf(m, __shfl_xor(m, off));
      float sum = 0.f;
      #pragma unroll
      for (int t = 0; t < 16; ++t) { vals[t] = __expf(vals[t] - m); sum += vals[t]; }
      #pragma unroll
      for (int off = 16; off > 0; off >>= 1) sum += __shfl_xor(sum, off);
      float inv = 1.f / sum;
      #pragma unroll
      for (int t = 0; t < 16; ++t) arow[lane + 32 * t] = (bf16_t)(vals[t] * inv);
    }
    __syncthreads();

    // GEMM2: O[j,c] = sum_i att[j,i]*qb[i,c]; wave -> (j-tile = w>>1, c-tile = ct)
    {
      int jtl = w >> 1;
      const bf16_t* attb = (const bf16_t*)S;         // row stride = 1024 bf16 elems
      v8f acc = {};
      #pragma unroll
      for (int s2 = 0; s2 < 16; ++s2) {
        const bf16_t* ap = attb + (jtl * 16 + nn) * 1024 + s2 * 32 + h * 8;
        v8bf alo = *(const v8bf*)ap;
        v8bf ahi = *(const v8bf*)(ap + 16);
        v16bf afrag = __builtin_shufflevector(alo, ahi,
            0,1,2,3,4,5,6,7,8,9,10,11,12,13,14,15);
        acc = __builtin_amdgcn_wmma_f32_16x16x32_bf16(
            false, afrag, false, bfrag2[s2], (short)0, acc, false, false);
      }
      int grow = jbase + jtl * 16 + h * 8;
      #pragma unroll
      for (int r = 0; r < 8; ++r)
        ol[(grow + r) * C_ + ct * 16 + nn] = acc[r];
    }
    __syncthreads();
  }

  // ---- min/max renorm per channel over 512 positions ----
  {
    float mn = 3.4e38f, mx = -3.4e38f;
    for (int i = w * 64; i < w * 64 + 64; ++i) {
      float v = ol[i * C_ + lane];
      mn = fminf(mn, v); mx = fmaxf(mx, v);
    }
    red[w * 32 + lane] = mn; red2[w * 32 + lane] = mx;
  }
  __syncthreads();
  if (tid < 32) {
    float mn = 3.4e38f, mx = -3.4e38f;
    for (int s2 = 0; s2 < 8; ++s2) {
      mn = fminf(mn, red[s2 * 32 + tid]); mx = fmaxf(mx, red2[s2 * 32 + tid]);
    }
    mnv[tid] = mn; scv[tid] = 1.f / (mx - mn + 1e-4f);
  }
  __syncthreads();

  // ---- faithful (C,n)->(hk,wk,dk,C) reinterpretation + Wp projection + scatter ----
  int bh = g >> 4, bw = (g >> 2) & 3, bd = g & 3;
  for (int p = tid; p < N_; p += 256) {
    float vs[32];
    #pragma unroll
    for (int cc = 0; cc < 32; ++cc) {
      int f = p * 32 + cc;
      int c = f >> 9, jj = f & 511;
      vs[cc] = (ol[jj * C_ + c] - mnv[c]) * scv[c];
    }
    int hh = p >> 6, ww = (p >> 3) & 7, dd = p & 7;
    int H = bh * 8 + hh, W = bw * 8 + ww, D = bd * 8 + dd;
    size_t base = ((size_t)bb * 32) * 32768 + (size_t)H * 1024 + (size_t)W * 32 + D;
    #pragma unroll
    for (int e = 0; e < 32; ++e) {
      float r = bps[e];
      #pragma unroll
      for (int cc = 0; cc < 32; ++cc) r += vs[cc] * wp[cc * 32 + e];
      out[base + (size_t)e * 32768] = r;   // (b, C, hq, wq, dq)
    }
  }
}

extern "C" void kernel_launch(void* const* d_in, const int* in_sizes, int n_in,
                              void* d_out, int out_size, void* d_ws, size_t ws_size,
                              hipStream_t stream) {
  (void)in_sizes; (void)n_in; (void)out_size; (void)ws_size;
  const float* q  = (const float*)d_in[0];
  const float* k  = (const float*)d_in[1];
  const float* Wq = (const float*)d_in[2];
  const float* bq = (const float*)d_in[3];
  const float* Wk = (const float*)d_in[4];
  const float* bk = (const float*)d_in[5];
  const float* Wp = (const float*)d_in[6];
  const float* bp = (const float*)d_in[7];
  float* out = (float*)d_out;

  // workspace: kf_bf16 (2*512*32 = 64KB), qb_bf16 (2*64*512*32 = 4MB)
  bf16_t* kf_ws = (bf16_t*)d_ws;
  bf16_t* qb_ws = (bf16_t*)((char*)d_ws + 65536);

  kf_proj_kernel<<<B_, 512, 0, stream>>>(k, Wk, bk, kf_ws);
  qb_proj_kernel<<<(B_ * NQ) / 256, 256, 0, stream>>>(q, Wq, bq, qb_ws);
  attn_kernel<<<B_ * G_, 256, SMEM_BYTES, stream>>>(kf_ws, qb_ws, Wp, bp, out);
}
